// MultiHeadAttention_55207509623384
// MI455X (gfx1250) — matrix operand order repaired and checked by reading, not compiled, not run on hardware
//
#include <hip/hip_runtime.h>

// MHA prefill for MI455X (gfx1250, wave32, WMMA + double-buffered async-LDS).
// B=2, S=2048, D=1024, H=16, DK=64.
// d_out layout (floats, concatenated return order):
//   output  [B,S,D]      at 0            (4,194,304)
//   weights [B,H,S,S]    at 4,194,304    (134,217,728)
//   K       [B,H,S,DK]   at 138,412,032  (4,194,304)
//   V       [B,H,S,DK]   at 142,606,336  (4,194,304)

typedef __attribute__((ext_vector_type(16))) __bf16 v16bf;
typedef __attribute__((ext_vector_type(8)))  __bf16 v8bf;
typedef __attribute__((ext_vector_type(4)))  __bf16 v4bf;
typedef __attribute__((ext_vector_type(8)))  float  v8f;

#define BQ   2
#define SQ   2048
#define DM   1024
#define HN   16
#define DKH  64

#define WTS_OFF   4194304ll
#define KV_K_OFF  138412032ll
#define KV_V_OFF  142606336ll

// ws layout in __bf16 elements
#define WS_XQ   0ll
#define WS_XK   4194304ll
#define WS_XV   8388608ll
#define WS_WQ   12582912ll
#define WS_WK   13631488ll
#define WS_WV   14680064ll
#define WS_WO   15728640ll
#define WS_QB   16777216ll
#define WS_KB   20971520ll
#define WS_VB   25165824ll
#define WS_AT   29360128ll   // total 33,554,432 bf16 = 64 MB

__device__ inline v4bf cvt4(const float4 f) {
    v4bf h = { (__bf16)f.x, (__bf16)f.y, (__bf16)f.z, (__bf16)f.w };
    return h;
}

// ---------------------------------------------------------------------------
// CDNA5 async copy (ASYNCcnt-tracked), SADDR form: SGPR64 base + i32 VGPR
// byte offset.  LDS byte offset = low 32 bits of the flat LDS address.
// ---------------------------------------------------------------------------
__device__ inline void async_ld_b128_s(unsigned long long sbase,
                                       unsigned int voff, unsigned int ldsoff) {
    asm volatile("global_load_async_to_lds_b128 %0, %1, %2"
                 :: "v"(ldsoff), "v"(voff), "s"(sbase) : "memory");
}
__device__ inline void wait_async_le5() {
    asm volatile("s_wait_asynccnt 5" ::: "memory");
}
__device__ inline void wait_async0() {
    asm volatile("s_wait_asynccnt 0" ::: "memory");
}
__device__ inline unsigned int lds32(const void* p) {
    return (unsigned int)(unsigned long long)p;
}

// ---------------------------------------------------------------------------
// fp32 -> bf16 elementwise (float4 -> 8B bf16x4 stores)
// ---------------------------------------------------------------------------
__global__ void cvt_bf16_kernel(const float* __restrict__ src,
                                __bf16* __restrict__ dst, int n4)
{
    int i = blockIdx.x * blockDim.x + threadIdx.x;
    if (i < n4)
        ((v4bf*)dst)[i] = cvt4(((const float4*)src)[i]);
}

// ---------------------------------------------------------------------------
// Per-wave 16x64 GEMM strip, bf16 operands, double-buffered async staging.
// A: bf16 row-major [M x K]; B: bf16 row-major [N x K] (X @ W^T form).
// As[2][16][32] m-major, Bs[2][4][16][32] n-major: fragments = contiguous
// 16B runs -> ds_load_b128 pairs straight into v_wmma.
// ---------------------------------------------------------------------------
__device__ inline void gemm_strip_async_db(const __bf16* __restrict__ A, int lda, int m0,
                                           const __bf16* __restrict__ B, int ldb, int n0,
                                           int K,
                                           __bf16 (*As)[16][32],
                                           __bf16 (*Bs)[4][16][32],
                                           int lane, v8f acc[4])
{
    const int r   = lane >> 1;           // tile row 0..15
    const int c16 = (lane & 1) * 16;     // 16-elem (32B) half of the row
    const int row = lane & 15;
    const int kbA = (lane < 16) ? 0 : 8;
    const int kbB = (lane < 16) ? 0 : 8;

    const unsigned long long aBase = (unsigned long long)A;
    const unsigned long long bBase = (unsigned long long)B;
    const unsigned int aOff = (unsigned int)((((long)(m0 + r) * lda) + c16) * 2);
    unsigned int bOff[4];
#pragma unroll
    for (int t = 0; t < 4; ++t)
        bOff[t] = (unsigned int)((((long)(n0 + t * 16 + r) * ldb) + c16) * 2);

    unsigned int ldsA[2], ldsB[2][4];
#pragma unroll
    for (int s = 0; s < 2; ++s) {
        ldsA[s] = lds32(&As[s][r][c16]);
#pragma unroll
        for (int t = 0; t < 4; ++t)
            ldsB[s][t] = lds32(&Bs[s][t][r][c16]);
    }

    // prologue: stage 0 -> buffer 0
    async_ld_b128_s(aBase, aOff, ldsA[0]);
#pragma unroll
    for (int t = 0; t < 4; ++t)
        async_ld_b128_s(bBase, bOff[t], ldsB[0][t]);

    const int nk = K >> 5;
    for (int i = 0; i < nk; ++i) {
        const int cur = i & 1;
        if (i + 1 < nk) {
            const unsigned int adv = (unsigned int)(i + 1) * 64u;  // 32 bf16 = 64B
            async_ld_b128_s(aBase, aOff + adv, ldsA[cur ^ 1]);
#pragma unroll
            for (int t = 0; t < 4; ++t)
                async_ld_b128_s(bBase, bOff[t] + adv, ldsB[cur ^ 1][t]);
            wait_async_le5();       // in-order completion: stage i has landed
        } else {
            wait_async0();
        }
        __syncthreads();

        const v8bf alo = *(const v8bf*)&As[cur][row][kbA];
        const v8bf ahi = *(const v8bf*)&As[cur][row][kbA + 16];
        const v16bf a = __builtin_shufflevector(alo, ahi,
            0, 1, 2, 3, 4, 5, 6, 7, 8, 9, 10, 11, 12, 13, 14, 15);
#pragma unroll
        for (int t = 0; t < 4; ++t) {
            const v8bf blo = *(const v8bf*)&Bs[cur][t][row][kbB];
            const v8bf bhi = *(const v8bf*)&Bs[cur][t][row][kbB + 16];
            const v16bf b = __builtin_shufflevector(blo, bhi,
                0, 1, 2, 3, 4, 5, 6, 7, 8, 9, 10, 11, 12, 13, 14, 15);
            acc[t] = __builtin_amdgcn_wmma_f32_16x16x32_bf16(
                false, a, false, b, (short)0, acc[t], false, false);
        }
        __syncthreads();            // WAR: buffer reused two stages later
    }
}

// ---------------------------------------------------------------------------
// fp32-source strip (cvt staging) for the attn@V GEMM whose operands are
// produced in fp32. B source row-major [K x N].
// ---------------------------------------------------------------------------
__device__ inline void gemm_strip_f32src(const float* __restrict__ A, int lda, int m0,
                                         const float* __restrict__ B, int ldb, int n0,
                                         int K,
                                         __bf16 (*As)[32], __bf16 (*Bs)[16][32],
                                         int lane, v8f acc[4])
{
    const int ar  = lane >> 3;
    const int ac  = (lane & 7) * 4;
    const int row = lane & 15;
    const int kbA = (lane < 16) ? 0 : 8;
    const int kbB = (lane < 16) ? 0 : 8;

    for (int k0 = 0; k0 < K; k0 += 32) {
#pragma unroll
        for (int it = 0; it < 4; ++it) {
            const int r = it * 4 + ar;
            const float4 f = *(const float4*)&A[(long)(m0 + r) * lda + k0 + ac];
            *(v4bf*)&As[r][ac] = cvt4(f);
        }
#pragma unroll
        for (int t = 0; t < 4; ++t)
#pragma unroll
            for (int it = 0; it < 4; ++it) {
                const int k  = it * 8 + (lane >> 2);
                const int n4 = (lane & 3) * 4;
                const float4 f =
                    *(const float4*)&B[(long)(k0 + k) * ldb + n0 + t * 16 + n4];
                Bs[t][n4 + 0][k] = (__bf16)f.x;
                Bs[t][n4 + 1][k] = (__bf16)f.y;
                Bs[t][n4 + 2][k] = (__bf16)f.z;
                Bs[t][n4 + 3][k] = (__bf16)f.w;
            }
        __syncthreads();

        const v8bf alo = *(const v8bf*)&As[row][kbA];
        const v8bf ahi = *(const v8bf*)&As[row][kbA + 16];
        const v16bf a = __builtin_shufflevector(alo, ahi,
            0, 1, 2, 3, 4, 5, 6, 7, 8, 9, 10, 11, 12, 13, 14, 15);
#pragma unroll
        for (int t = 0; t < 4; ++t) {
            const v8bf blo = *(const v8bf*)&Bs[t][row][kbB];
            const v8bf bhi = *(const v8bf*)&Bs[t][row][kbB + 16];
            const v16bf b = __builtin_shufflevector(blo, bhi,
                0, 1, 2, 3, 4, 5, 6, 7, 8, 9, 10, 11, 12, 13, 14, 15);
            acc[t] = __builtin_amdgcn_wmma_f32_16x16x32_bf16(
                false, a, false, b, (short)0, acc[t], false, false);
        }
        __syncthreads();
    }
}

// ---------------------------------------------------------------------------
// 1) QKV projections on bf16 operands.  grid = (M/16, N/256, 3), block = 128.
//    bf16 Q/K/V -> ws; fp32 K/V -> their output slots.
// ---------------------------------------------------------------------------
__global__ void qkv_kernel(const __bf16* __restrict__ Xq, const __bf16* __restrict__ Xk,
                           const __bf16* __restrict__ Xv,
                           const __bf16* __restrict__ Wqb, const __bf16* __restrict__ Wkb,
                           const __bf16* __restrict__ Wvb,
                           const float* __restrict__ bq, const float* __restrict__ bk,
                           const float* __restrict__ bv,
                           __bf16* __restrict__ Qb, __bf16* __restrict__ Kb,
                           __bf16* __restrict__ Vb, float* __restrict__ out)
{
    __shared__ __bf16 As[4][2][16][32];
    __shared__ __bf16 Bs[4][2][4][16][32];
    const int lane = threadIdx.x & 31;
    const int w    = threadIdx.x >> 5;
    const int m0   = blockIdx.x * 16;
    const int n0   = (blockIdx.y * 4 + w) * 64;
    const int sel  = blockIdx.z;

    const __bf16* X    = sel == 0 ? Xq  : (sel == 1 ? Xk  : Xv);
    const __bf16* Wm   = sel == 0 ? Wqb : (sel == 1 ? Wkb : Wvb);
    const float*  bias = sel == 0 ? bq  : (sel == 1 ? bk  : bv);
    __bf16* bdst = sel == 0 ? Qb : (sel == 1 ? Kb : Vb);
    float*  fdst = sel == 1 ? out + KV_K_OFF : out + KV_V_OFF;   // unused for sel==0

    v8f acc[4] = {};
    gemm_strip_async_db(X, DM, m0, Wm, DM, n0, DM, As[w], Bs[w], lane, acc);

#pragma unroll
    for (int t = 0; t < 4; ++t) {
        const int n  = n0 + t * 16 + (lane & 15);
        const int h  = n >> 6, dk = n & 63;
        const float bn = bias[n];
#pragma unroll
        for (int v = 0; v < 8; ++v) {
            int m = m0 + v + ((lane < 16) ? 0 : 8);
            int b = m >> 11, s = m & 2047;
            long idx = ((long)(b * HN + h) * SQ + s) * DKH + dk;
            float val = acc[t][v] + bn;
            bdst[idx] = (__bf16)val;
            if (sel != 0) fdst[idx] = val;
        }
    }
}

// ---------------------------------------------------------------------------
// 2) scores = Qb @ Kb^T * 1/8 + mask -> weights region (fp32).
//    grid = (S/16, S/256, B*H), block = 128.
// ---------------------------------------------------------------------------
__global__ void scores_kernel(const __bf16* __restrict__ Qb,
                              const __bf16* __restrict__ Kb,
                              const float* __restrict__ mask,
                              float* __restrict__ out)
{
    __shared__ __bf16 As[4][2][16][32];
    __shared__ __bf16 Bs[4][2][4][16][32];
    const int lane = threadIdx.x & 31;
    const int w    = threadIdx.x >> 5;
    const int m0   = blockIdx.x * 16;
    const int n0   = (blockIdx.y * 4 + w) * 64;
    const int bh   = blockIdx.z;

    const __bf16* Qh = Qb + (long)bh * SQ * DKH;
    const __bf16* Kh = Kb + (long)bh * SQ * DKH;
    float*        Wt = out + WTS_OFF + (long)bh * SQ * SQ;

    v8f acc[4] = {};
    gemm_strip_async_db(Qh, DKH, m0, Kh, DKH, n0, DKH, As[w], Bs[w], lane, acc);

    const float scale = 0.125f;
#pragma unroll
    for (int t = 0; t < 4; ++t) {
        const int n = n0 + t * 16 + (lane & 15);
#pragma unroll
        for (int v = 0; v < 8; ++v) {
            int m = m0 + v + ((lane < 16) ? 0 : 8);
            Wt[(long)m * SQ + n] = acc[t][v] * scale + mask[(long)m * SQ + n];
        }
    }
}

// ---------------------------------------------------------------------------
// 3) In-place row softmax over 2048 cols (float4 vectorized).
// ---------------------------------------------------------------------------
__global__ void softmax_kernel(float* __restrict__ wts)
{
    float4* p4 = (float4*)(wts + (long)blockIdx.x * SQ);
    const int t = threadIdx.x;
    __shared__ float red[256];

    float4 a0 = p4[t];
    float4 a1 = p4[t + 256];
    float mx = fmaxf(fmaxf(fmaxf(a0.x, a0.y), fmaxf(a0.z, a0.w)),
                     fmaxf(fmaxf(a1.x, a1.y), fmaxf(a1.z, a1.w)));
    red[t] = mx; __syncthreads();
    for (int s = 128; s > 0; s >>= 1) {
        if (t < s) red[t] = fmaxf(red[t], red[t + s]);
        __syncthreads();
    }
    mx = red[0]; __syncthreads();

    a0.x = __expf(a0.x - mx); a0.y = __expf(a0.y - mx);
    a0.z = __expf(a0.z - mx); a0.w = __expf(a0.w - mx);
    a1.x = __expf(a1.x - mx); a1.y = __expf(a1.y - mx);
    a1.z = __expf(a1.z - mx); a1.w = __expf(a1.w - mx);
    float sum = (a0.x + a0.y + a0.z + a0.w) + (a1.x + a1.y + a1.z + a1.w);
    red[t] = sum; __syncthreads();
    for (int s = 128; s > 0; s >>= 1) {
        if (t < s) red[t] += red[t + s];
        __syncthreads();
    }
    const float inv = 1.0f / red[0];
    a0.x *= inv; a0.y *= inv; a0.z *= inv; a0.w *= inv;
    a1.x *= inv; a1.y *= inv; a1.z *= inv; a1.w *= inv;
    p4[t] = a0;
    p4[t + 256] = a1;
}

// ---------------------------------------------------------------------------
// 4) attn = weights @ V -> bf16 attn in ws ([B,S,D]).
//    grid = (S/128, 1, B*H), block = 256 (wave w: rows bx*128+w*16, DK strip).
// ---------------------------------------------------------------------------
__global__ void attnv_kernel(const float* __restrict__ out_buf,
                             __bf16* __restrict__ attnb)
{
    __shared__ __bf16 As[8][16][32];
    __shared__ __bf16 Bs[8][4][16][32];
    const int lane = threadIdx.x & 31;
    const int w    = threadIdx.x >> 5;
    const int bh   = blockIdx.z;
    const int m0   = blockIdx.x * 128 + w * 16;

    const float* Wt = out_buf + WTS_OFF + (long)bh * SQ * SQ;
    const float* Vh = out_buf + KV_V_OFF + (long)bh * SQ * DKH;

    v8f acc[4] = {};
    gemm_strip_f32src(Wt, SQ, m0, Vh, DKH, 0, SQ, As[w], Bs[w], lane, acc);

    const int b = bh >> 4, h = bh & 15;
#pragma unroll
    for (int t = 0; t < 4; ++t) {
        const int n = t * 16 + (lane & 15);
#pragma unroll
        for (int v = 0; v < 8; ++v) {
            int m = m0 + v + ((lane < 16) ? 0 : 8);
            attnb[((long)(b * SQ + m)) * DM + h * DKH + n] = (__bf16)acc[t][v];
        }
    }
}

// ---------------------------------------------------------------------------
// 5) output = attnb @ Wo^T + bo (async bf16 path). grid = (M/16, N/256),
//    block = 128.
// ---------------------------------------------------------------------------
__global__ void oproj_kernel(const __bf16* __restrict__ attnb,
                             const __bf16* __restrict__ Wob,
                             const float* __restrict__ bo,
                             float* __restrict__ out)
{
    __shared__ __bf16 As[4][2][16][32];
    __shared__ __bf16 Bs[4][2][4][16][32];
    const int lane = threadIdx.x & 31;
    const int w    = threadIdx.x >> 5;
    const int m0   = blockIdx.x * 16;
    const int n0   = (blockIdx.y * 4 + w) * 64;

    v8f acc[4] = {};
    gemm_strip_async_db(attnb, DM, m0, Wob, DM, n0, DM, As[w], Bs[w], lane, acc);

#pragma unroll
    for (int t = 0; t < 4; ++t) {
        const int n = n0 + t * 16 + (lane & 15);
        const float bn = bo[n];
#pragma unroll
        for (int v = 0; v < 8; ++v) {
            int m = m0 + v + ((lane < 16) ? 0 : 8);
            out[(long)m * DM + n] = acc[t][v] + bn;
        }
    }
}

// ---------------------------------------------------------------------------
extern "C" void kernel_launch(void* const* d_in, const int* in_sizes, int n_in,
                              void* d_out, int out_size, void* d_ws, size_t ws_size,
                              hipStream_t stream)
{
    (void)in_sizes; (void)n_in; (void)out_size; (void)ws_size;

    const float* query = (const float*)d_in[0];
    const float* key   = (const float*)d_in[1];
    const float* value = (const float*)d_in[2];
    const float* mask  = (const float*)d_in[3];
    const float* Wq    = (const float*)d_in[4];
    const float* bq    = (const float*)d_in[5];
    const float* Wk    = (const float*)d_in[6];
    const float* bk    = (const float*)d_in[7];
    const float* Wv    = (const float*)d_in[8];
    const float* bv    = (const float*)d_in[9];
    const float* Wo    = (const float*)d_in[10];
    const float* bo    = (const float*)d_in[11];

    float*  out = (float*)d_out;
    __bf16* wsb = (__bf16*)d_ws;

    dim3 b256(256);
    dim3 b128(128);

    // 0) one-time fp32 -> bf16 operand conversion
    cvt_bf16_kernel<<<dim3(4096), b256, 0, stream>>>(query, wsb + WS_XQ, 1048576);
    cvt_bf16_kernel<<<dim3(4096), b256, 0, stream>>>(key,   wsb + WS_XK, 1048576);
    cvt_bf16_kernel<<<dim3(4096), b256, 0, stream>>>(value, wsb + WS_XV, 1048576);
    cvt_bf16_kernel<<<dim3(1024), b256, 0, stream>>>(Wq,    wsb + WS_WQ, 262144);
    cvt_bf16_kernel<<<dim3(1024), b256, 0, stream>>>(Wk,    wsb + WS_WK, 262144);
    cvt_bf16_kernel<<<dim3(1024), b256, 0, stream>>>(Wv,    wsb + WS_WV, 262144);
    cvt_bf16_kernel<<<dim3(1024), b256, 0, stream>>>(Wo,    wsb + WS_WO, 262144);

    // 1) Q/K/V projections (double-buffered async bf16 tiles)
    qkv_kernel<<<dim3(256, 4, 3), b128, 0, stream>>>(
        wsb + WS_XQ, wsb + WS_XK, wsb + WS_XV,
        wsb + WS_WQ, wsb + WS_WK, wsb + WS_WV,
        bq, bk, bv,
        wsb + WS_QB, wsb + WS_KB, wsb + WS_VB, out);

    // 2) scaled+masked scores
    scores_kernel<<<dim3(128, 8, 32), b128, 0, stream>>>(
        wsb + WS_QB, wsb + WS_KB, mask, out);

    // 3) in-place softmax over each of B*H*S rows
    softmax_kernel<<<dim3(65536), b256, 0, stream>>>(out + WTS_OFF);

    // 4) attn = weights @ V (fp32 sources, cvt staging) -> bf16 attn
    attnv_kernel<<<dim3(16, 1, 32), b256, 0, stream>>>(out, wsb + WS_AT);

    // 5) output projection (double-buffered async bf16 path)
    oproj_kernel<<<dim3(256, 4, 1), b128, 0, stream>>>(
        wsb + WS_AT, wsb + WS_WO, bo, out);
}
